// _TemplateTriAtt_20873541058562
// MI455X (gfx1250) — compile-verified, matrix operand order checked by reading, not run on hardware
//
#include <hip/hip_runtime.h>
#include <hip/hip_bf16.h>

#define NN 256
#define DD 64
#define HH 4
#define DHD 32
#define INNER 128
#define SCALE_F 0.17677669529663687f  // 1/sqrt(32)

typedef __attribute__((ext_vector_type(16))) __bf16 v16bf;
typedef __attribute__((ext_vector_type(8)))  float  v8f;
typedef __attribute__((ext_vector_type(4)))  int    v4i;

#define HAS_ASYNC_LDS __has_builtin(__builtin_amdgcn_global_load_async_to_lds_b128)

// K-index held by element e of a 16-bit 16x32 A (or 32x16 B) fragment.
// Lanes 0-15: K in {0..7, 16..23}; lanes 16-31: K in {8..15, 24..31}.
// Elements 0..7 and 8..15 are each contiguous runs of 8 -> 16B vector loads.
__device__ __forceinline__ int frag_k(int e, int lane) {
  return ((e & 8) << 1) + ((lane & 16) >> 1) + (e & 7);
}

__device__ __forceinline__ v8f wmma_bf16(v16bf a, v16bf b, v8f c) {
  return __builtin_amdgcn_wmma_f32_16x16x32_bf16(false, a, false, b, (short)0, c,
                                                 false, false);
}

__device__ __forceinline__ float fast_rcp(float x) {
  return __builtin_amdgcn_rcpf(x);  // v_rcp_f32, no Newton chain
}

__device__ __forceinline__ float hsum32(float v) {
#pragma unroll
  for (int m = 1; m <= 16; m <<= 1) v += __shfl_xor(v, m, 32);
  return v;
}
__device__ __forceinline__ float hsum16(float v) {
#pragma unroll
  for (int m = 1; m <= 8; m <<= 1) v += __shfl_xor(v, m, 32);
  return v;
}
__device__ __forceinline__ float hmax16(float v) {
#pragma unroll
  for (int m = 1; m <= 8; m <<= 1) v = fmaxf(v, __shfl_xor(v, m, 32));
  return v;
}

// bias stored in WMMA-C-fragment-tiled order:
//   bias_t[(((h*16 + it)*16 + jt)*32 + lanefrag)*8 + vv]
// lanefrag = (j&15) | ((i&8)<<1), vv = i&7, it = i>>4, jt = j>>4.
__device__ __forceinline__ size_t bias_t_addr(int hh, int i, int j) {
  int it = i >> 4, jt = j >> 4;
  int lanefrag = (j & 15) | ((i & 8) << 1);
  int vv = i & 7;
  return ((size_t)(((hh * 16 + it) * 16 + jt) * 32 + lanefrag) << 3) + vv;
}

// ---------------------------------------------------------------------------
// Kernel 1: layernorm + QKV / gate / bias projections.
// Grid: 128 blocks x 256 threads; each wave loops 4 position-tiles so the
// 66KB LDS weight image is amortized 4x.
// V is written TRANSPOSED [n][h][d][j] so kernel 2's B-fragments vectorize.
// ---------------------------------------------------------------------------
__global__ __launch_bounds__(256) void tri_proj(
    const float* __restrict__ z, const float* __restrict__ ln_w,
    const float* __restrict__ ln_b, const float* __restrict__ w_qkv,
    const float* __restrict__ w_bias, const float* __restrict__ w_gate,
    __bf16* __restrict__ q_ws, __bf16* __restrict__ k_ws,
    __bf16* __restrict__ v_ws, __bf16* __restrict__ gate_ws,
    float* __restrict__ bias_t) {
  extern __shared__ char smem[];
  __bf16* s_w  = (__bf16*)smem;                          // 512 x 64 bf16
  float*  s_wb = (float*)(smem + 512 * 64 * 2);          // 4 x 64 f32
  __bf16* s_h  = (__bf16*)(smem + 512 * 64 * 2 + 1024);  // 8 waves x 16 x 64

  const int tid = threadIdx.x, lane = tid & 31, wave = tid >> 5;

  for (int idx = tid; idx < 512 * 64; idx += 256)
    s_w[idx] = (__bf16)((idx < 384 * 64) ? w_qkv[idx] : w_gate[idx - 384 * 64]);
  if (tid < 256) s_wb[tid] = w_bias[tid];
  __syncthreads();

  __bf16* hrow = s_h + wave * (16 * 64);
  const float lw0 = ln_w[lane], lw1 = ln_w[lane + 32];
  const float lb0 = ln_b[lane], lb1 = ln_b[lane + 32];
  const int cc = lane & 15;
  const int rsel = (lane & 16) >> 1;

#pragma unroll 1
  for (int rep = 0; rep < 4; ++rep) {
    const int p0 = blockIdx.x * 512 + rep * 128 + wave * 16;
    const int n0 = p0 >> 8;         // all 16 rows of a tile share one n
    const int ij0 = (p0 & 255) + rsel;

    // LayerNorm, 16 rows per wave, lane-parallel over 64 channels.
#pragma unroll 1
    for (int r = 0; r < 16; ++r) {
      const float* zp = z + (size_t)(p0 + r) * DD;
      if (r + 2 < 16) __builtin_prefetch(z + (size_t)(p0 + r + 2) * DD, 0, 1);
      float x0 = zp[lane], x1 = zp[lane + 32];
      float s  = hsum32(x0 + x1);
      float ss = hsum32(x0 * x0 + x1 * x1);
      float mu = s * (1.f / 64.f);
      float var = ss * (1.f / 64.f) - mu * mu;
      float rstd = __frsqrt_rn(var + 1e-5f);
      hrow[r * 64 + lane]      = (__bf16)((x0 - mu) * rstd * lw0 + lb0);
      hrow[r * 64 + lane + 32] = (__bf16)((x1 - mu) * rstd * lw1 + lb1);
    }

    // bias = h @ w_bias^T (width 4) -> stored in C-fragment-tiled layout
#pragma unroll 1
    for (int hh = 0; hh < HH; ++hh) {
#pragma unroll 1
      for (int r = 0; r < 16; ++r) {
        float v = (float)hrow[r * 64 + lane] * s_wb[hh * 64 + lane] +
                  (float)hrow[r * 64 + lane + 32] * s_wb[hh * 64 + lane + 32];
        v = hsum32(v);
        int pos = p0 + r;
        if (lane == 0) bias_t[bias_t_addr(hh, pos >> 8, pos & 255)] = v;
      }
    }

    // A fragments: h tile, two K=32 chunks.
    v16bf a0, a1;
#pragma unroll
    for (int e = 0; e < 16; ++e) {
      a0[e] = hrow[cc * 64 + frag_k(e, lane)];
      a1[e] = hrow[cc * 64 + 32 + frag_k(e, lane)];
    }

#pragma unroll 1
    for (int t = 0; t < 32; ++t) {  // 24 qkv tiles + 8 gate tiles
      const int obase = t * 16;
      v16bf b0, b1;
#pragma unroll
      for (int e = 0; e < 16; ++e) {
        b0[e] = s_w[(obase + cc) * 64 + frag_k(e, lane)];
        b1[e] = s_w[(obase + cc) * 64 + 32 + frag_k(e, lane)];
      }
      v8f c = {};
      c = wmma_bf16(a0, b0, c);
      c = wmma_bf16(a1, b1, c);

      if (obase < 384) {
        const int which = obase / 128, hh = (obase % 128) / 32, dbase = obase % 32;
        if (which < 2) {  // Q, K: [n][h][j][d]; addr = base + vv*DHD
          __bf16* dst = ((which == 0) ? q_ws : k_ws) +
                        ((size_t)((n0 * HH + hh) * NN + ij0) * DHD + dbase + cc);
#pragma unroll
          for (int vv = 0; vv < 8; ++vv) dst[vv * DHD] = (__bf16)c[vv];
        } else {  // V transposed [n][h][d][j]; addr = base + vv
          __bf16* dst = v_ws +
                        ((size_t)((n0 * HH + hh) * DHD + dbase + cc) * NN + ij0);
#pragma unroll
          for (int vv = 0; vv < 8; ++vv) dst[vv] = (__bf16)c[vv];
        }
      } else {
        const int e0 = obase - 384;
        __bf16* dst = gate_ws + ((size_t)(p0 + rsel) * INNER + e0 + cc);
#pragma unroll
        for (int vv = 0; vv < 8; ++vv)
          dst[vv * INNER] = (__bf16)fast_rcp(1.f + __expf(-c[vv]));  // sigmoid
      }
    }
  }  // rep
}

// ---------------------------------------------------------------------------
// Kernel 2: per-n attention (4 heads) + gate + output projection.
// Grid: 256 blocks x 256 threads. K/V (128KB) staged memory->LDS via the
// CDNA5 async-LDS path (ASYNCcnt), else VGPR copies.
// ---------------------------------------------------------------------------
__global__ __launch_bounds__(256) void tri_attn(
    const __bf16* __restrict__ q_ws, const __bf16* __restrict__ k_ws,
    const __bf16* __restrict__ v_ws, const __bf16* __restrict__ gate_ws,
    const float* __restrict__ bias_t, const unsigned char* __restrict__ mask,
    const float* __restrict__ w_out, float* __restrict__ out) {
  extern __shared__ char smem[];
  __bf16* s_k    = (__bf16*)smem;                // [H][256][32] bf16 = 64KB
  __bf16* s_v    = (__bf16*)(smem + 65536);      // [H][32][256] bf16 = 64KB
  __bf16* s_wout = (__bf16*)(smem + 131072);     // [64][128] bf16 = 16KB
  float*  s_mask = (float*)(smem + 147456);      // 256 f32
  __bf16* s_p    = (__bf16*)(smem + 148480);     // 8 waves x 16x32 bf16
  __bf16* s_o    = (__bf16*)(smem + 156672);     // 8 waves x 16x128 bf16

  const int tid = threadIdx.x, lane = tid & 31, wave = tid >> 5;
  const int n = blockIdx.x;

  {  // stage K and V (64KB each)
    const char* gk = (const char*)(k_ws + (size_t)n * (HH * NN * DHD));
    const char* gv = (const char*)(v_ws + (size_t)n * (HH * NN * DHD));
#if HAS_ASYNC_LDS
    typedef __attribute__((address_space(1))) v4i* gp128;
    typedef __attribute__((address_space(3))) v4i* lp128;
    for (int off = tid * 16; off < HH * NN * DHD * 2; off += 256 * 16) {
      __builtin_amdgcn_global_load_async_to_lds_b128(
          (gp128)(gk + off), (lp128)((char*)s_k + off), 0, 0);
      __builtin_amdgcn_global_load_async_to_lds_b128(
          (gp128)(gv + off), (lp128)((char*)s_v + off), 0, 0);
    }
#else
    const uint4* srck = (const uint4*)gk;
    const uint4* srcv = (const uint4*)gv;
    uint4* dk = (uint4*)s_k;
    uint4* dv = (uint4*)s_v;
    for (int idx = tid; idx < (HH * NN * DHD) / 8; idx += 256) {
      dk[idx] = srck[idx];
      dv[idx] = srcv[idx];
    }
#endif
  }
  for (int idx = tid; idx < DD * INNER; idx += 256) s_wout[idx] = (__bf16)w_out[idx];
  if (tid < 256) s_mask[tid] = mask[n * NN + tid] ? 0.f : -1.0e30f;
#if HAS_ASYNC_LDS
#if __has_builtin(__builtin_amdgcn_s_wait_asynccnt)
  __builtin_amdgcn_s_wait_asynccnt(0);
#else
  asm volatile("s_wait_asynccnt 0" ::: "memory");
#endif
#endif
  __syncthreads();

  __bf16* pst = s_p + wave * (16 * 32);
  __bf16* ost = s_o + wave * (16 * 128);
  const int cc = lane & 15;
  const int rsel = (lane & 16) >> 1;

#pragma unroll 1
  for (int it = wave; it < 16; it += 8) {
    const int ibase = it * 16;

#pragma unroll 1
    for (int hh = 0; hh < HH; ++hh) {
      // bias tile base for this (head, i-tile): lane-contiguous fragments
      const float* bt = bias_t + ((size_t)((hh * 16 + it) * 16) * 32 << 3);
      __builtin_prefetch(bt + ((size_t)lane << 3), 0, 1);

      // Q A-fragment (DH = 32 -> exactly one WMMA K-chunk)
      v16bf aq;
      const __bf16* qb = q_ws + (size_t)(n * HH + hh) * NN * DHD;
#pragma unroll
      for (int e = 0; e < 16; ++e)
        aq[e] = qb[(ibase + cc) * DHD + frag_k(e, lane)];

      // S = Q K^T : 16 j-tiles, kept in registers for softmax
      v8f s[16];
#pragma unroll
      for (int jt = 0; jt < 16; ++jt) {
        v16bf bk;
#pragma unroll
        for (int e = 0; e < 16; ++e)
          bk[e] = s_k[(hh * NN + jt * 16 + cc) * DHD + frag_k(e, lane)];
        v8f cz = {};
        s[jt] = wmma_bf16(aq, bk, cz);
      }

      // fused scale + pair-bias + mask; bias fragment-tiled: 8 consecutive
      // f32 per lane per j-tile -> global_load_b128 x2
#pragma unroll
      for (int jt = 0; jt < 16; ++jt) {
        const float* bb = bt + ((size_t)(jt * 32 + lane) << 3);
        const float ma = s_mask[jt * 16 + cc];
#pragma unroll
        for (int vv = 0; vv < 8; ++vv)
          s[jt][vv] = s[jt][vv] * SCALE_F + bb[vv] + ma;
      }

      // row softmax (rows split across 16-lane halves -> half-wave reduce)
      v8f mrow;
#pragma unroll
      for (int vv = 0; vv < 8; ++vv) {
        float m = s[0][vv];
#pragma unroll
        for (int jt = 1; jt < 16; ++jt) m = fmaxf(m, s[jt][vv]);
        mrow[vv] = hmax16(m);
      }
      v8f rsum = {};
#pragma unroll
      for (int jt = 0; jt < 16; ++jt) {
#pragma unroll
        for (int vv = 0; vv < 8; ++vv) {
          float p = __expf(s[jt][vv] - mrow[vv]);
          s[jt][vv] = p;
          rsum[vv] += p;
        }
      }
      v8f rinv;
#pragma unroll
      for (int vv = 0; vv < 8; ++vv) rinv[vv] = fast_rcp(hsum16(rsum[vv]));

      // O = P V : re-layout P (C-frag -> A-frag) through per-wave LDS tile
      v8f o0 = {}, o1 = {};
#pragma unroll
      for (int jc = 0; jc < 8; ++jc) {
#pragma unroll
        for (int half = 0; half < 2; ++half) {
#pragma unroll
          for (int vv = 0; vv < 8; ++vv)
            pst[(vv + rsel) * 32 + half * 16 + cc] = (__bf16)s[jc * 2 + half][vv];
        }
        v16bf ap;
#pragma unroll
        for (int e = 0; e < 16; ++e) ap[e] = pst[cc * 32 + frag_k(e, lane)];
        // V transposed in LDS: contiguous in fragment element -> ds_load_b128
        v16bf bv0, bv1;
#pragma unroll
        for (int e = 0; e < 16; ++e) {
          bv0[e] = s_v[(hh * DHD + cc) * NN + jc * 32 + frag_k(e, lane)];
          bv1[e] = s_v[(hh * DHD + 16 + cc) * NN + jc * 32 + frag_k(e, lane)];
        }
        o0 = wmma_bf16(ap, bv0, o0);
        o1 = wmma_bf16(ap, bv1, o1);
      }

      // normalized, gated O into per-wave LDS (A-source for out-proj)
      const __bf16* gp =
          gate_ws + ((size_t)(n * NN + ibase + rsel) * INNER + hh * DHD);
#pragma unroll
      for (int vv = 0; vv < 8; ++vv) {
        float g0 = (float)gp[vv * INNER + cc];
        float g1 = (float)gp[vv * INNER + 16 + cc];
        ost[(vv + rsel) * 128 + hh * DHD + cc]      = (__bf16)(o0[vv] * rinv[vv] * g0);
        ost[(vv + rsel) * 128 + hh * DHD + 16 + cc] = (__bf16)(o1[vv] * rinv[vv] * g1);
      }
    }  // heads

    // out = (O*gate) @ Wout^T : 16x128 x 128x64
    v16bf ao[4];
#pragma unroll
    for (int kc = 0; kc < 4; ++kc)
#pragma unroll
      for (int e = 0; e < 16; ++e)
        ao[kc][e] = ost[cc * 128 + kc * 32 + frag_k(e, lane)];

#pragma unroll 1
    for (int dt = 0; dt < 4; ++dt) {
      v8f c = {};
#pragma unroll
      for (int kc = 0; kc < 4; ++kc) {
        v16bf bw;
#pragma unroll
        for (int e = 0; e < 16; ++e)
          bw[e] = s_wout[(dt * 16 + cc) * INNER + kc * 32 + frag_k(e, lane)];
        c = wmma_bf16(ao[kc], bw, c);
      }
      float* op = out + ((size_t)(n * NN + ibase + rsel) * DD + dt * 16 + cc);
#pragma unroll
      for (int vv = 0; vv < 8; ++vv) op[vv * DD] = c[vv];
    }
  }
}

// ---------------------------------------------------------------------------
extern "C" void kernel_launch(void* const* d_in, const int* in_sizes, int n_in,
                              void* d_out, int out_size, void* d_ws,
                              size_t ws_size, hipStream_t stream) {
  const float* z            = (const float*)d_in[0];
  const unsigned char* mask = (const unsigned char*)d_in[1];  // jnp.bool_
  const float* ln_w         = (const float*)d_in[2];
  const float* ln_b         = (const float*)d_in[3];
  const float* w_qkv        = (const float*)d_in[4];
  const float* w_bias       = (const float*)d_in[5];
  const float* w_out        = (const float*)d_in[6];
  const float* w_gate       = (const float*)d_in[7];

  // Workspace: q/k/v/gate bf16 (4 x 16.8MB) + bias f32 (1MB) ~= 68MB.
  const size_t NQ = (size_t)NN * NN * INNER;
  char* ws = (char*)d_ws;
  __bf16* q_ws = (__bf16*)ws;
  __bf16* k_ws = q_ws + NQ;
  __bf16* v_ws = k_ws + NQ;
  __bf16* g_ws = v_ws + NQ;
  float* bias_t = (float*)(ws + 4 * NQ * sizeof(__bf16));

  const size_t smem1 = 512 * 64 * 2 + 1024 + 8 * 16 * 64 * 2;  // ~82KB
  const size_t smem2 = 189440;                                  // ~185KB
  tri_proj<<<(NN * NN) / 512, 256, smem1, stream>>>(
      z, ln_w, ln_b, w_qkv, w_bias, w_gate, q_ws, k_ws, v_ws, g_ws, bias_t);
  tri_attn<<<NN, 256, smem2, stream>>>(q_ws, k_ws, v_ws, g_ws, bias_t, mask,
                                       w_out, (float*)d_out);
}
